// lossEvaluator_47347719471166
// MI455X (gfx1250) — matrix operand order repaired and checked by reading, not compiled, not run on hardware
//
#include <hip/hip_runtime.h>

typedef __attribute__((ext_vector_type(16))) __bf16 v16bf;
typedef __attribute__((ext_vector_type(16))) float  v16f;
typedef __attribute__((ext_vector_type(8)))  float  v8f;
typedef __attribute__((ext_vector_type(4)))  float  v4f;

#define NB     256   // batch (images == texts)
#define NP     64    // patches per image
#define NF     1024  // feature dim
#define MARGIN 0.1f

// Monotonic uint encoding of float for atomic max.
__device__ __forceinline__ unsigned fkey(float f) {
  unsigned b = __float_as_uint(f);
  return (b & 0x80000000u) ? ~b : (b | 0x80000000u);
}
__device__ __forceinline__ float funkey(unsigned k) {
  unsigned b = (k & 0x80000000u) ? (k & 0x7fffffffu) : ~k;
  return __uint_as_float(b);
}

// Vector-native fp32 -> bf16 hi/lo split: hi = RNE(f); lo = RNE(f - hi).
__device__ __forceinline__ void split16(const v16f& af, v16bf& hi, v16bf& lo) {
  hi = __builtin_convertvector(af, v16bf);
  v16f lof;
#pragma unroll
  for (int e = 0; e < 16; ++e) lof[e] = af[e] - (float)hi[e];
  lo = __builtin_convertvector(lof, v16bf);
}

// ---------------------------------------------------------------------------
// Kernel 0: one-time fp32 -> bf16 hi/lo split of disFtr (256x1024 = 256K elts).
// Layout preserved ([n][k] row-major) so GEMM B-fragments are contiguous loads.
// ---------------------------------------------------------------------------
__global__ __launch_bounds__(256)
void split_dis_kernel(const float* __restrict__ disFtr,
                      __bf16* __restrict__ disHi,
                      __bf16* __restrict__ disLo) {
  const int base = blockIdx.x * blockDim.x + threadIdx.x;
#pragma unroll
  for (int r = 0; r < 4; ++r) {
    const int i = base + r * (NB * NF / 4);
    float  f = disFtr[i];
    __bf16 h = (__bf16)f;
    disHi[i] = h;
    disLo[i] = (__bf16)(f - (float)h);
  }
}

// ---------------------------------------------------------------------------
// Kernel 1: one workgroup (128 threads = 4 waves) per image b.
// Wave w owns ALL 4 M-tiles x N-tiles 4w..4w+3 -> 16 acc tiles.
// FP32 emulated via bf16 hi/lo split: A*B ~= Ahi*Bhi + Ahi*Blo + Alo*Bhi.
// Cooperative A split: per K-step wave w converts only M-tile w and shares
// the packed hi/lo fragments through LDS (4x less VALU, A traffic moved off
// the vector-memory path). Each B hi/lo pair feeds 12 WMMAs.
// ---------------------------------------------------------------------------
__global__ __launch_bounds__(128)
void simmax_kernel(const float* __restrict__ imFtr,
                   const __bf16* __restrict__ disHi,
                   const __bf16* __restrict__ disLo,
                   float* __restrict__ simMax) {
  const int b    = blockIdx.x;
  const int tid  = threadIdx.x;
  const int wave = tid >> 5;   // 0..3 -> N group (tiles 4w..4w+3); also M-tile it splits
  const int lane = tid & 31;
  const int l16  = lane & 15;
  const int lhi  = lane >> 4;

  // A fragment (16x32 bf16, MxK): lane holds row M=l16 (of tile mt);
  // element e -> K = kk + lhi*16 + e  (16 contiguous floats).
  // This wave loads/splits only M-tile == wave.
  const float* aRow0 =
      imFtr + ((size_t)b * NP + (size_t)(wave * 16 + l16)) * NF + lhi * 16;

  // B fragment (32x16 bf16, KxN): lane holds column N=l16;
  // element e -> K = kk + lhi*16 + e  (16 contiguous bf16 = one v16bf load).
  const size_t bOff = (size_t)lhi * 16;

  // LDS staging for the 4 A fragment pairs: [tile][hi/lo][lane][16 bf16] = 8KB.
  __shared__ __align__(16) __bf16 aLds[4][2][32][16];

  const v8f vzero = {0.f, 0.f, 0.f, 0.f, 0.f, 0.f, 0.f, 0.f};
  v8f acc[16];  // [mt*4 + nt]
#pragma unroll
  for (int i = 0; i < 16; ++i) acc[i] = vzero;

  for (int kk = 0; kk < NF; kk += 32) {
    // ---- this wave: load + split its M-tile, publish to LDS ----
    {
      v4f t0 = *(const v4f*)(aRow0 + kk);
      v4f t1 = *(const v4f*)(aRow0 + kk + 4);
      v4f t2 = *(const v4f*)(aRow0 + kk + 8);
      v4f t3 = *(const v4f*)(aRow0 + kk + 12);
      v16f af;
#pragma unroll
      for (int e = 0; e < 4; ++e) {
        af[e]      = t0[e];
        af[e + 4]  = t1[e];
        af[e + 8]  = t2[e];
        af[e + 12] = t3[e];
      }
      v16bf hi, lo;
      split16(af, hi, lo);
      *(v16bf*)&aLds[wave][0][lane][0] = hi;
      *(v16bf*)&aLds[wave][1][lane][0] = lo;
    }
    __syncthreads();  // fragments visible to all 4 waves

    // ---- gather all 4 A fragment pairs from LDS ----
    v16bf ahi[4], alo[4];
#pragma unroll
    for (int mt = 0; mt < 4; ++mt) {
      ahi[mt] = *(const v16bf*)&aLds[mt][0][lane][0];
      alo[mt] = *(const v16bf*)&aLds[mt][1][lane][0];
    }

#pragma unroll
    for (int nt = 0; nt < 4; ++nt) {
      const int    n  = (wave * 4 + nt) * 16 + l16;
      const size_t ro = (size_t)n * NF + kk + bOff;
      v16bf bhi = *(const v16bf*)(disHi + ro);   // 32B contiguous (2x b128)
      v16bf blo = *(const v16bf*)(disLo + ro);

      // 12 WMMAs per B fragment pair: 4 M-tiles x 3 split terms.
#pragma unroll
      for (int mt = 0; mt < 4; ++mt) {
        v8f c = acc[mt * 4 + nt];
        c = __builtin_amdgcn_wmma_f32_16x16x32_bf16(
            false, ahi[mt], false, bhi, (short)0, c, false, false);
        c = __builtin_amdgcn_wmma_f32_16x16x32_bf16(
            false, ahi[mt], false, blo, (short)0, c, false, false);
        c = __builtin_amdgcn_wmma_f32_16x16x32_bf16(
            false, alo[mt], false, bhi, (short)0, c, false, false);
        acc[mt * 4 + nt] = c;
      }
    }
    __syncthreads();  // all reads done before next K-step overwrites LDS
  }

  // ---- max over the 64 patch rows -> simMax[b][0..255] ----
  // C/D layout: lane holds N=l16; element r -> row M = mt*16 + r + 8*lhi.
  // Each lane reduces 32 rows in-register; lanes L and L+16 combine via LDS.
  __shared__ unsigned smax[NB];
  smax[tid]       = 0u;  // encoded minimum
  smax[tid + 128] = 0u;
  __syncthreads();
#pragma unroll
  for (int nt = 0; nt < 4; ++nt) {
    float m = acc[nt][0];
#pragma unroll
    for (int mt = 0; mt < 4; ++mt)
#pragma unroll
      for (int r = 0; r < 8; ++r) m = fmaxf(m, acc[mt * 4 + nt][r]);
    const int n = (wave * 4 + nt) * 16 + l16;
    atomicMax(&smax[n], fkey(m));
  }
  __syncthreads();
  simMax[(size_t)b * NB + tid]       = funkey(smax[tid]);
  simMax[(size_t)b * NB + tid + 128] = funkey(smax[tid + 128]);
}

// ---------------------------------------------------------------------------
// Kernel 2: fold both ranking losses in one pass over simMax using the
// symmetric negative-pair mask:
//   loss+lossBi = sum_{i,j} mask_ij*(relu(s_ij-pos_i+m)+relu(s_ij-pos_j+m)) / pairNum
// ---------------------------------------------------------------------------
__global__ __launch_bounds__(256)
void loss_kernel(const float* __restrict__ simMax,
                 const int* __restrict__ labels,
                 float* __restrict__ out) {
  __shared__ float pos[NB];
  __shared__ int   lbl[NB];
  __shared__ float rs[NB];
  __shared__ float rc[NB];
  const int t = threadIdx.x;
  pos[t] = simMax[(size_t)t * NB + t];
  lbl[t] = labels[t];
  __syncthreads();

  float       s  = 0.f, c = 0.f;
  const float pt = pos[t];
  const int   lt = lbl[t];
  for (int i = 0; i < NB; ++i) {
    float v = simMax[(size_t)i * NB + t];  // column t, row i
    if (lbl[i] != lt) {
      s += fmaxf(v - pos[i] + MARGIN, 0.f) + fmaxf(v - pt + MARGIN, 0.f);
      c += 1.f;
    }
  }
  rs[t] = s;
  rc[t] = c;
  __syncthreads();
  for (int off = 128; off > 0; off >>= 1) {
    if (t < off) {
      rs[t] += rs[t + off];
      rc[t] += rc[t + off];
    }
    __syncthreads();
  }
  if (t == 0) out[0] = rs[0] / rc[0];
}

extern "C" void kernel_launch(void* const* d_in, const int* in_sizes, int n_in,
                              void* d_out, int out_size, void* d_ws, size_t ws_size,
                              hipStream_t stream) {
  const float* imFtr  = (const float*)d_in[0];  // [256,64,1024] f32
  const float* disFtr = (const float*)d_in[1];  // [256,1024]    f32
  const int*   lbl    = (const int*)d_in[2];    // [256]

  // Workspace layout (1.25 MB total):
  //   [0,       256KB) simMax  f32 [256][256]
  //   [256KB,   768KB) disHi   bf16 [256][1024]
  //   [768KB, 1.25MB)  disLo   bf16 [256][1024]
  float*  simMax = (float*)d_ws;
  __bf16* disHi  = (__bf16*)((char*)d_ws + 256 * 1024);
  __bf16* disLo  = (__bf16*)((char*)d_ws + 768 * 1024);

  split_dis_kernel<<<NB, 256, 0, stream>>>(disFtr, disHi, disLo);
  simmax_kernel<<<NB, 128, 0, stream>>>(imFtr, disHi, disLo, simMax);
  loss_kernel<<<1, 256, 0, stream>>>(simMax, lbl, (float*)d_out);
}
